// Encoder_31945966747775
// MI455X (gfx1250) — matrix operand order repaired and checked by reading, not compiled, not run on hardware
//
#include <hip/hip_runtime.h>
#include <hip/hip_bf16.h>
#include <cstdint>

// ---------------------------------------------------------------- constants
constexpr int B  = 16;
constexpr int N  = 512;
constexpr int D  = 256;
constexpr int H  = 8;
constexpr int L  = 4;
constexpr int DK = 32;

constexpr float EPS_LN   = 1e-6f;
constexpr float EPS_ADJ  = 1e-6f;
constexpr float kNegInf  = -1.0e12f;
constexpr float LAM_ATT  = 0.3f;
constexpr float LAM_DIST = 0.3f;
constexpr float LAM_ADJ  = 0.4f;
constexpr float SLOPE    = 0.1f;
constexpr float INV_SQRT_DK = 0.17677669529663687f; // 1/sqrt(32)

typedef __attribute__((ext_vector_type(16))) _Float16 v16h;
typedef __attribute__((ext_vector_type(8)))  float    v8f;

// ---------------------------------------------------------------- WMMA utils
__device__ inline v8f wmma_f16(v16h a, v16h b, v8f c) {
  // D = A(16x32 f16) * B(32x16 f16) + C(16x16 f32)
  return __builtin_amdgcn_wmma_f32_16x16x32_f16(false, a, false, b,
                                                (short)0, c, false, false);
}

// A fragment: 16x32 (MxK) f16, row-major source with leading dim lda.
// lane: m = lane&15, g = lane>>4. VGPRs 0-3 hold K = g*8 + 0..7,
// VGPRs 4-7 hold K = 16 + g*8 + 0..7 (contiguous 8-half runs -> 2x 16B loads).
__device__ inline v16h load_a_frag(const _Float16* Abase, int lda, int m0, int k0) {
  int lane = threadIdx.x & 31;
  int m = m0 + (lane & 15);
  int g = lane >> 4;
  const _Float16* p = Abase + (size_t)m * lda + k0;
  union { v16h v; float4 q[2]; } u;
  u.q[0] = *(const float4*)(p + g * 8);
  u.q[1] = *(const float4*)(p + 16 + g * 8);
  return u.v;
}

// B fragment: 32x16 (KxN) f16 supplied TRANSPOSED (Bt[n][k], leading dim ldb).
// lane: n = lane&15, g = lane>>4; halves 0-7 hold K = g*8 + 0..7, halves 8-15 hold K = 16 + g*8 + 0..7.
__device__ inline v16h load_b_frag(const _Float16* Bt, int ldb, int n0, int k0) {
  int lane = threadIdx.x & 31;
  int n = n0 + (lane & 15);
  int g = lane >> 4;
  const _Float16* p = Bt + (size_t)n * ldb + k0;
  union { v16h v; float4 q[2]; } u;
  u.q[0] = *(const float4*)(p + g * 8);
  u.q[1] = *(const float4*)(p + 16 + g * 8);
  return u.v;
}

// Async global->LDS copy (CDNA5 TDM-adjacent path, ASYNCcnt-tracked).
// Generic LDS pointers carry the LDS byte offset in their low 32 bits.
__device__ inline void async_copy_b128_to_lds(void* ldsDst, const void* gSrc) {
  uint32_t lds = (uint32_t)(uintptr_t)ldsDst;
  asm volatile("global_load_async_to_lds_b128 %0, %1, off"
               :: "v"(lds), "v"(gSrc) : "memory");
}
__device__ inline void wait_asynccnt0() {
  asm volatile("s_wait_asynccnt 0x0" ::: "memory");
}

// ---------------------------------------------------------------- p_mix
// pmix[b,i,j] = LAM_DIST*softmax_j(mask? -dist : -inf) + LAM_ADJ*adj/rowsum
__global__ void pmix_kernel(const float* __restrict__ adj,
                            const float* __restrict__ dist,
                            const int*   __restrict__ mask,
                            float*       __restrict__ pmix) {
  __shared__ float r0[256];
  __shared__ float r1[256];
  int row = blockIdx.x;             // b*N + i
  int b   = row >> 9;
  int tid = threadIdx.x;            // 256 threads, 2 keys each
  const float* dr = dist + (size_t)row * N;
  const float* ar = adj  + (size_t)row * N;
  const int*   mr = mask + b * N;

  float d0 = mr[tid]       ? -dr[tid]       : -3.0e38f;
  float d1 = mr[tid + 256] ? -dr[tid + 256] : -3.0e38f;
  float a0 = ar[tid], a1 = ar[tid + 256];
  r0[tid] = fmaxf(d0, d1);
  r1[tid] = a0 + a1;
  __syncthreads();
  for (int off = 128; off > 0; off >>= 1) {
    if (tid < off) {
      r0[tid] = fmaxf(r0[tid], r0[tid + off]);
      r1[tid] += r1[tid + off];
    }
    __syncthreads();
  }
  float mx   = r0[0];
  float asum = r1[0];
  __syncthreads();
  float e0 = (d0 > -1.0e37f) ? __expf(d0 - mx) : 0.f;
  float e1 = (d1 > -1.0e37f) ? __expf(d1 - mx) : 0.f;
  r0[tid] = e0 + e1;
  __syncthreads();
  for (int off = 128; off > 0; off >>= 1) {
    if (tid < off) r0[tid] += r0[tid + off];
    __syncthreads();
  }
  float pinv = 1.0f / r0[0];
  float ainv = 1.0f / (asum + EPS_ADJ);
  pmix[(size_t)row * N + tid]       = LAM_DIST * e0 * pinv + LAM_ADJ * a0 * ainv;
  pmix[(size_t)row * N + tid + 256] = LAM_DIST * e1 * pinv + LAM_ADJ * a1 * ainv;
}

// ---------------------------------------------------------------- weight prep
// Wt[l][n][k] = (f16) W[l][k][n]   (D=256)
__global__ void wt_kernel(const float* __restrict__ W, _Float16* __restrict__ Wt) {
  int idx = blockIdx.x * 256 + threadIdx.x;     // over L*D*D
  int l = idx >> 16;
  int r = idx & 65535;
  int n = r >> 8, k = r & 255;
  Wt[idx] = (_Float16)W[((size_t)l << 16) + (k << 8) + n];
}

// ---------------------------------------------------------------- LayerNorm
// ddof=1 std; one row per block, 256 threads (one element each).
__global__ void ln_kernel(const float* __restrict__ x,
                          const float* __restrict__ ga,
                          const float* __restrict__ gb,
                          _Float16* __restrict__ out16,
                          float*    __restrict__ out32) {
  __shared__ float rs[256];
  __shared__ float rq[256];
  int row = blockIdx.x, tid = threadIdx.x;
  float v = x[(size_t)row * D + tid];
  rs[tid] = v;
  rq[tid] = v * v;
  __syncthreads();
  for (int off = 128; off > 0; off >>= 1) {
    if (tid < off) { rs[tid] += rs[tid + off]; rq[tid] += rq[tid + off]; }
    __syncthreads();
  }
  float mean = rs[0] * (1.0f / D);
  float var  = (rq[0] - (float)D * mean * mean) * (1.0f / (D - 1));
  float sd   = sqrtf(fmaxf(var, 0.0f));
  float y    = ga[tid] * (v - mean) / (sd + EPS_LN) + gb[tid];
  if (out16) out16[(size_t)row * D + tid] = (_Float16)y;
  else       out32[(size_t)row * D + tid] = y;
}

// ---------------------------------------------------------------- fused QKV
// 2 M-tiles per wave, 3 weight matrices -> 6 WMMAs per 5 fragment loads.
// q,k -> [B,H,N,DK] f16 ; v -> [B,H,DK,N] f16 (pre-transposed B operand).
__global__ void qkv_kernel(const _Float16* __restrict__ h16,
                           const _Float16* __restrict__ Wqt,
                           const _Float16* __restrict__ Wkt,
                           const _Float16* __restrict__ Wvt,
                           const float* __restrict__ bq,
                           const float* __restrict__ bk,
                           const float* __restrict__ bv,
                           _Float16* __restrict__ q16,
                           _Float16* __restrict__ k16,
                           _Float16* __restrict__ v16t) {
  int wave = threadIdx.x >> 5, lane = threadIdx.x & 31;
  int t  = blockIdx.x * 8 + wave;       // tile id over (M/32)x(D/16)
  int tm = (t >> 4) * 32;               // 2 M-tiles of 16
  int tn = (t & 15) * 16;
  v8f q0 = {}, q1 = {}, k0a = {}, k1a = {}, v0 = {}, v1 = {};
#pragma unroll
  for (int k0 = 0; k0 < D; k0 += 32) {
    v16h a0 = load_a_frag(h16, D, tm, k0);
    v16h a1 = load_a_frag(h16, D, tm + 16, k0);
    v16h wq = load_b_frag(Wqt, D, tn, k0);
    v16h wk = load_b_frag(Wkt, D, tn, k0);
    v16h wv = load_b_frag(Wvt, D, tn, k0);
    q0  = wmma_f16(a0, wq, q0);
    q1  = wmma_f16(a1, wq, q1);
    k0a = wmma_f16(a0, wk, k0a);
    k1a = wmma_f16(a1, wk, k1a);
    v0  = wmma_f16(a0, wv, v0);
    v1  = wmma_f16(a1, wv, v1);
  }
  int g = lane >> 4;
  int n = tn + (lane & 15);
  int hh = n >> 5, dk = n & 31;
  float bqv = bq[n], bkv = bk[n], bvv = bv[n];
  auto stQKV = [&](v8f aq, v8f ak, v8f av, int m0) {
#pragma unroll
    for (int r = 0; r < 8; ++r) {
      int m = m0 + r + 8 * g;           // token in [0, B*N)
      int bb = m >> 9, nn = m & 511;
      size_t qi = ((size_t)((bb * H + hh) * N) + nn) * DK + dk;
      q16[qi] = (_Float16)(aq[r] + bqv);
      k16[qi] = (_Float16)(ak[r] + bkv);
      size_t vi = ((size_t)((bb * H + hh) * DK) + dk) * N + nn;
      v16t[vi] = (_Float16)(av[r] + bvv);
    }
  };
  stQKV(q0, k0a, v0, tm);
  stQKV(q1, k1a, v1, tm + 16);
}

// ---------------------------------------------------------------- attention
// One block per (b,h,qtile16): scores via WMMA, masked softmax, mix with
// p_mix, P*V via WMMA with cross-wave partial reduction.
__global__ void attn_kernel(const _Float16* __restrict__ q16,
                            const _Float16* __restrict__ k16,
                            const _Float16* __restrict__ v16t,
                            const float*    __restrict__ pmix,
                            const int*      __restrict__ mask,
                            _Float16*       __restrict__ att16) {
  __shared__ __attribute__((aligned(16))) float    SF[16 * N];   // scores, then partials
  __shared__ __attribute__((aligned(16))) _Float16 P16[16 * N];  // probabilities f16
  __shared__ __attribute__((aligned(16))) _Float16 Qs[16 * DK];  // q tile
  __shared__ float Red[256];

  int tid  = threadIdx.x;
  int wave = tid >> 5, lane = tid & 31;
  int blk = blockIdx.x;
  int qt = blk & 31;            // N/16 = 32 query tiles
  int hh = (blk >> 5) & 7;
  int bb = blk >> 8;

  const _Float16* qhead = q16  + ((size_t)(bb * H + hh)) * N * DK;
  const _Float16* khead = k16  + ((size_t)(bb * H + hh)) * N * DK;
  const _Float16* vhead = v16t + ((size_t)(bb * H + hh)) * DK * N;

  // --- stage Q tile (16x32 halves = 1024 B) via async global->LDS (64x16B)
  if (tid < 64) {
    async_copy_b128_to_lds((char*)Qs + tid * 16,
                           (const char*)(qhead + (size_t)qt * 16 * DK) + tid * 16);
  }
  wait_asynccnt0();
  __syncthreads();

  // --- scores: wave w covers key tiles 4w..4w+3 (one WMMA each, K=DK=32)
  {
    v16h a = load_a_frag(Qs, DK, 0, 0);
    for (int kt = wave * 4; kt < wave * 4 + 4; ++kt) {
      v8f acc = {};
      acc = wmma_f16(a, load_b_frag(khead, DK, kt * 16, 0), acc);
      int g = lane >> 4;
      int key = kt * 16 + (lane & 15);
      int mk = mask[bb * N + key];
#pragma unroll
      for (int r = 0; r < 8; ++r) {
        int row = r + 8 * g;
        float s = acc[r] * INV_SQRT_DK;
        SF[row * N + key] = (mk != 0) ? s : kNegInf;
      }
    }
  }
  __syncthreads();

  // --- softmax + mixing: 16 threads per query row
  {
    int row = tid >> 4, j = tid & 15;
    float mx = -3.0e38f;
    for (int key = j; key < N; key += 16) mx = fmaxf(mx, SF[row * N + key]);
    Red[tid] = mx;
    __syncthreads();
    float rmx = -3.0e38f;
    for (int jj = 0; jj < 16; ++jj) rmx = fmaxf(rmx, Red[row * 16 + jj]);
    __syncthreads();
    float sum = 0.f;
    for (int key = j; key < N; key += 16) {
      float e = __expf(SF[row * N + key] - rmx);
      SF[row * N + key] = e;
      sum += e;
    }
    Red[tid] = sum;
    __syncthreads();
    float rsum = 0.f;
    for (int jj = 0; jj < 16; ++jj) rsum += Red[row * 16 + jj];
    float inv = 1.0f / rsum;
    const float* pm = pmix + ((size_t)(bb * N + qt * 16 + row)) * N;
    for (int key = j; key < N; key += 16) {
      float pv = LAM_ATT * (SF[row * N + key] * inv) + pm[key];
      P16[row * N + key] = (_Float16)pv;
    }
  }
  __syncthreads();     // SF free after this point; reused for partials

  // --- att = P(16xN) * V(NxDK): wave w takes k-steps 2w,2w+1 (32 keys each)
  {
    for (int nt = 0; nt < 2; ++nt) {
      v8f acc = {};
      for (int s = wave * 2; s < wave * 2 + 2; ++s) {
        v16h a = load_a_frag(P16, N, 0, s * 32);
        acc = wmma_f16(a, load_b_frag(vhead, N, nt * 16, s * 32), acc);
      }
      int g = lane >> 4, n = nt * 16 + (lane & 15);
#pragma unroll
      for (int r = 0; r < 8; ++r) {
        int row = r + 8 * g;
        SF[(wave * 16 + row) * 32 + n] = acc[r];   // partials [8][16][32]
      }
    }
  }
  __syncthreads();

  // --- reduce 8 wave partials, scatter back to [token][h*32+dk]
  {
    for (int idx = tid; idx < 16 * 32; idx += 256) {
      int row = idx >> 5, dk = idx & 31;
      float s = 0.f;
      for (int w = 0; w < 8; ++w) s += SF[(w * 16 + row) * 32 + dk];
      int token = bb * N + qt * 16 + row;
      att16[(size_t)token * D + hh * DK + dk] = (_Float16)s;
    }
  }
}

// ---------------------------------------------------------------- generic GEMM
// C(MxD) = A(MxD,f16) * W(DxD via Wt) + bias. 2x2 register blocking per wave:
// 4 WMMAs per 2 A + 2 B fragment loads. Epilogue MODE:
//   1: Cf32 += val            (Wo projection residual)
//   2: Cf16  = leaky(val)     (FFN1)
//   3: Cf32 += leaky(val)     (FFN2 residual)
template <int MODE>
__global__ void gemm_kernel(const _Float16* __restrict__ A,
                            const _Float16* __restrict__ Bt,
                            const float* __restrict__ bias,
                            float* __restrict__ Cf32,
                            _Float16* __restrict__ Cf16) {
  int wave = threadIdx.x >> 5, lane = threadIdx.x & 31;
  int t  = blockIdx.x * 8 + wave;   // super-tile over (M/32)x(D/32)
  int tm = (t >> 3) * 32;           // D/32 == 8 n super-tiles
  int tn = (t & 7) * 32;
  v8f c00 = {}, c01 = {}, c10 = {}, c11 = {};
#pragma unroll
  for (int k0 = 0; k0 < D; k0 += 32) {
    v16h a0 = load_a_frag(A, D, tm, k0);
    v16h a1 = load_a_frag(A, D, tm + 16, k0);
    v16h b0 = load_b_frag(Bt, D, tn, k0);
    v16h b1 = load_b_frag(Bt, D, tn + 16, k0);
    c00 = wmma_f16(a0, b0, c00);
    c01 = wmma_f16(a0, b1, c01);
    c10 = wmma_f16(a1, b0, c10);
    c11 = wmma_f16(a1, b1, c11);
  }
  int g = lane >> 4;
  int ln = lane & 15;
  auto store = [&](v8f acc, int m0, int n0) {
    int n = n0 + ln;
    float bv = bias[n];
#pragma unroll
    for (int r = 0; r < 8; ++r) {
      int m = m0 + r + 8 * g;
      float val = acc[r] + bv;
      if (MODE == 1) {
        Cf32[(size_t)m * D + n] += val;
      } else if (MODE == 2) {
        Cf16[(size_t)m * D + n] = (_Float16)(val > 0.f ? val : SLOPE * val);
      } else {
        float lv = val > 0.f ? val : SLOPE * val;
        Cf32[(size_t)m * D + n] += lv;
      }
    }
  };
  store(c00, tm, tn);
  store(c01, tm, tn + 16);
  store(c10, tm + 16, tn);
  store(c11, tm + 16, tn + 16);
}

// ---------------------------------------------------------------- launcher
extern "C" void kernel_launch(void* const* d_in, const int* in_sizes, int n_in,
                              void* d_out, int out_size, void* d_ws, size_t ws_size,
                              hipStream_t stream) {
  (void)in_sizes; (void)n_in; (void)out_size; (void)ws_size;

  const float* x_in  = (const float*)d_in[0];
  const int*   mask  = (const int*)  d_in[1];
  const float* adj   = (const float*)d_in[2];
  const float* dist  = (const float*)d_in[3];
  // d_in[4] = edges_att (unused)
  const float* Wq = (const float*)d_in[5];   const float* bq = (const float*)d_in[6];
  const float* Wk = (const float*)d_in[7];   const float* bk = (const float*)d_in[8];
  const float* Wv = (const float*)d_in[9];   const float* bv = (const float*)d_in[10];
  const float* Wo = (const float*)d_in[11];  const float* bo = (const float*)d_in[12];
  const float* Wf1 = (const float*)d_in[13]; const float* bf1 = (const float*)d_in[14];
  const float* Wf2 = (const float*)d_in[15]; const float* bf2 = (const float*)d_in[16];
  const float* ln1_a = (const float*)d_in[17]; const float* ln1_b = (const float*)d_in[18];
  const float* ln2_a = (const float*)d_in[19]; const float* ln2_b = (const float*)d_in[20];
  const float* lnf_a = (const float*)d_in[21]; const float* lnf_b = (const float*)d_in[22];
  float* out = (float*)d_out;

  // workspace carve-up
  char* p = (char*)d_ws;
  auto take = [&](size_t bytes) -> char* {
    char* r = p;
    p += (bytes + 255) & ~(size_t)255;
    return r;
  };
  float*    pmix  = (float*)   take((size_t)B * N * N * 4);   // 16 MB
  float*    x     = (float*)   take((size_t)B * N * D * 4);   //  8 MB
  _Float16* h16   = (_Float16*)take((size_t)B * N * D * 2);   //  4 MB
  _Float16* q16   = (_Float16*)take((size_t)B * H * N * DK * 2);
  _Float16* k16   = (_Float16*)take((size_t)B * H * N * DK * 2);
  _Float16* v16t  = (_Float16*)take((size_t)B * H * DK * N * 2);
  _Float16* att16 = (_Float16*)take((size_t)B * N * D * 2);
  _Float16* g16   = (_Float16*)take((size_t)B * N * D * 2);
  _Float16* Wqt  = (_Float16*)take((size_t)L * D * D * 2);
  _Float16* Wkt  = (_Float16*)take((size_t)L * D * D * 2);
  _Float16* Wvt  = (_Float16*)take((size_t)L * D * D * 2);
  _Float16* Wot  = (_Float16*)take((size_t)L * D * D * 2);
  _Float16* Wf1t = (_Float16*)take((size_t)L * D * D * 2);
  _Float16* Wf2t = (_Float16*)take((size_t)L * D * D * 2);

  // x working copy (residual stream)
  hipMemcpyAsync(x, x_in, (size_t)B * N * D * 4, hipMemcpyDeviceToDevice, stream);

  // one-time precompute
  pmix_kernel<<<B * N, 256, 0, stream>>>(adj, dist, mask, pmix);
  const int wtBlocks = L * D * D / 256;   // 1024
  wt_kernel<<<wtBlocks, 256, 0, stream>>>(Wq,  Wqt);
  wt_kernel<<<wtBlocks, 256, 0, stream>>>(Wk,  Wkt);
  wt_kernel<<<wtBlocks, 256, 0, stream>>>(Wv,  Wvt);
  wt_kernel<<<wtBlocks, 256, 0, stream>>>(Wo,  Wot);
  wt_kernel<<<wtBlocks, 256, 0, stream>>>(Wf1, Wf1t);
  wt_kernel<<<wtBlocks, 256, 0, stream>>>(Wf2, Wf2t);

  const int qkvBlocks  = (B * N / 32) * (D / 16) / 8;   // 512
  const int gemmBlocks = (B * N / 32) * (D / 32) / 8;   // 256
  const int attnBlocks = B * H * (N / 16);              // 4096
  const size_t WD = (size_t)D * D;

  for (int l = 0; l < L; ++l) {
    ln_kernel<<<B * N, 256, 0, stream>>>(x, ln1_a + l * D, ln1_b + l * D, h16, nullptr);
    qkv_kernel<<<qkvBlocks, 256, 0, stream>>>(h16, Wqt + l * WD, Wkt + l * WD, Wvt + l * WD,
                                              bq + l * D, bk + l * D, bv + l * D,
                                              q16, k16, v16t);
    attn_kernel<<<attnBlocks, 256, 0, stream>>>(q16, k16, v16t, pmix, mask, att16);
    gemm_kernel<1><<<gemmBlocks, 256, 0, stream>>>(att16, Wot + l * WD, bo + l * D, x, nullptr);
    ln_kernel<<<B * N, 256, 0, stream>>>(x, ln2_a + l * D, ln2_b + l * D, h16, nullptr);
    gemm_kernel<2><<<gemmBlocks, 256, 0, stream>>>(h16, Wf1t + l * WD, bf1 + l * D, nullptr, g16);
    gemm_kernel<3><<<gemmBlocks, 256, 0, stream>>>(g16, Wf2t + l * WD, bf2 + l * D, x, nullptr);
  }

  ln_kernel<<<B * N, 256, 0, stream>>>(x, lnf_a, lnf_b, nullptr, out);
}
